// LlamaAttention_26938034880666
// MI455X (gfx1250) — matrix operand order repaired and checked by reading, not compiled, not run on hardware
//
#include <hip/hip_runtime.h>
#include <hip/hip_bf16.h>

typedef __attribute__((ext_vector_type(16))) __bf16 v16bf;
typedef __attribute__((ext_vector_type(8)))  float  v8f;
typedef int v4i __attribute__((vector_size(16)));   // matches builtin's V4i param

#define Bc 2
#define Sc 2048
#define Dc 4096
#define Hc 32
#define KVHc 8
#define HDc 128
#define NEGC (-1.0e9f)

// -------- CDNA5 async load-to-LDS (guarded; fallback = register copy) -----
#if defined(__has_builtin)
# if __has_builtin(__builtin_amdgcn_global_load_async_to_lds_b128)
#  define HAVE_ASYNC_B128 1
# endif
# if __has_builtin(__builtin_amdgcn_s_wait_asynccnt)
#  define HAVE_WAIT_ASYNC 1
# endif
#endif

__device__ __forceinline__ void async_copy16(const __bf16* g, __bf16* l) {
#if defined(HAVE_ASYNC_B128)
  __builtin_amdgcn_global_load_async_to_lds_b128(
      (__attribute__((address_space(1))) v4i*)(void*)g,
      (__attribute__((address_space(3))) v4i*)(void*)l, 0, 0);
#else
  *reinterpret_cast<uint4*>(l) = *reinterpret_cast<const uint4*>(g);
#endif
}

__device__ __forceinline__ void wait_async() {
#if defined(HAVE_ASYNC_B128)
# if defined(HAVE_WAIT_ASYNC)
  __builtin_amdgcn_s_wait_asynccnt(0);
# else
  asm volatile("s_wait_asynccnt 0" ::: "memory");
# endif
#endif
}

// ---------------- WMMA fragment loaders (bf16, 16x16x32) ------------------
// A (16x32, MxK): lane L: row m = L%16, kb = (L<16)?0:8,
//   vgprs 0..3 = K kb..kb+7, vgprs 4..7 = K kb+16..kb+23  (two 16B chunks)
__device__ __forceinline__ v16bf frag_a(const __bf16* base, int ld, int lane) {
  int m  = lane & 15;
  int kb = (lane < 16) ? 0 : 8;
  const __bf16* p = base + (size_t)m * ld + kb;
  union { uint4 u[2]; v16bf v; } f;
  f.u[0] = *reinterpret_cast<const uint4*>(p);
  f.u[1] = *reinterpret_cast<const uint4*>(p + 16);
  return f.v;
}
// B (32x16, KxN): lane L: col n = L%16, kb = (L<16)?0:8, K kb..kb+7 and kb+16..kb+23.
// base laid out so element (k,n) lives at base[n*ld + k]
__device__ __forceinline__ v16bf frag_b(const __bf16* base, int ld, int lane) {
  int n  = lane & 15;
  int kb = (lane < 16) ? 0 : 8;
  const __bf16* p = base + (size_t)n * ld + kb;
  union { uint4 u[2]; v16bf v; } f;
  f.u[0] = *reinterpret_cast<const uint4*>(p);
  f.u[1] = *reinterpret_cast<const uint4*>(p + 16);
  return f.v;
}

__device__ __forceinline__ v8f wmma_bf16(v16bf a, v16bf b, v8f c) {
  return __builtin_amdgcn_wmma_f32_16x16x32_bf16(false, a, false, b,
                                                 (short)0, c, false, false);
}

// ---------------- one-pass f32 -> bf16 cast (n % 1024 == 0) ---------------
__global__ void __launch_bounds__(256)
cast_bf16_kernel(const float* __restrict__ in, __bf16* __restrict__ out) {
  size_t i = ((size_t)blockIdx.x * 256 + threadIdx.x) * 4;
  float4 v = *reinterpret_cast<const float4*>(in + i);
  union { __bf16 o[4]; uint2 u; } p;
  p.o[0] = (__bf16)v.x; p.o[1] = (__bf16)v.y;
  p.o[2] = (__bf16)v.z; p.o[3] = (__bf16)v.w;
  *reinterpret_cast<uint2*>(out + i) = p.u;
}

// ---------------- GEMM: C[M,N] = A[M,K] * W[N,K]^T (bf16 in) --------------
// Block: 256 thr (8 waves), tile 128x64, BK=64; LDS filled with async B128.
template <typename TC>
__global__ void __launch_bounds__(256)
gemm_wmma(const __bf16* __restrict__ A, const __bf16* __restrict__ W,
          TC* __restrict__ C, int M, int N, int K) {
  constexpr int BM = 128, BN = 64, BK = 64, LDA = BK + 8;
  __shared__ __align__(16) __bf16 As[BM][LDA];
  __shared__ __align__(16) __bf16 Ws[BN][LDA];

  const int tid  = threadIdx.x;
  const int lane = tid & 31;
  const int wid  = tid >> 5;           // 0..7
  const int wm   = wid >> 1;           // 0..3  (rows)
  const int wn   = wid & 1;            // 0..1  (cols)
  const int bm   = blockIdx.x * BM;
  const int bn   = blockIdx.y * BN;

  v8f acc[2][2] = {};

  for (int k0 = 0; k0 < K; k0 += BK) {
    // stage A (1024 x 16B chunks) and W (512 chunks) into LDS, async
#pragma unroll
    for (int i = 0; i < 4; ++i) {
      int idx = i * 256 + tid;
      int r = idx >> 3, c = (idx & 7) * 8;
      async_copy16(&A[(size_t)(bm + r) * K + k0 + c], &As[r][c]);
    }
#pragma unroll
    for (int i = 0; i < 2; ++i) {
      int idx = i * 256 + tid;
      int r = idx >> 3, c = (idx & 7) * 8;
      async_copy16(&W[(size_t)(bn + r) * K + k0 + c], &Ws[r][c]);
    }
    if (k0 + BK < K) {   // prefetch next tiles into GL2 (global_prefetch_b8)
      __builtin_prefetch(&A[(size_t)(bm + (tid >> 1)) * K + k0 + BK], 0, 1);
      __builtin_prefetch(&W[(size_t)(bn + (tid >> 2)) * K + k0 + BK], 0, 1);
    }
    wait_async();
    __syncthreads();

#pragma unroll
    for (int kk = 0; kk < BK; kk += 32) {
      v16bf af[2], bf[2];
#pragma unroll
      for (int i = 0; i < 2; ++i)
        af[i] = frag_a(&As[wm * 32 + i * 16][kk], LDA, lane);
#pragma unroll
      for (int j = 0; j < 2; ++j)
        bf[j] = frag_b(&Ws[wn * 32 + j * 16][kk], LDA, lane);
#pragma unroll
      for (int i = 0; i < 2; ++i)
#pragma unroll
        for (int j = 0; j < 2; ++j)
          acc[i][j] = wmma_bf16(af[i], bf[j], acc[i][j]);
    }
    __syncthreads();
  }

  // C 16x16 f32 layout: lane n = L%16, vgpr r -> row r + (L<16?0:8)
  const int hb = (lane & 16) ? 8 : 0;
#pragma unroll
  for (int i = 0; i < 2; ++i)
#pragma unroll
    for (int j = 0; j < 2; ++j) {
      int mb = bm + wm * 32 + i * 16 + hb;
      int nn = bn + wn * 32 + j * 16 + (lane & 15);
#pragma unroll
      for (int r = 0; r < 8; ++r)
        C[(size_t)(mb + r) * N + nn] = (TC)acc[i][j][r];
    }
}

// ---------------- RoPE + layout change (bf16 in, bf16 out) ----------------
// in : [B, S, nh*HD] bf16    out: [B, nh, S, HD] bf16
__global__ void __launch_bounds__(128)
rope_kernel(const __bf16* __restrict__ P, const float* __restrict__ cosb,
            const float* __restrict__ sinb, __bf16* __restrict__ Out,
            int nh, int apply_rope) {
  const int hd = threadIdx.x;            // 0..127
  const int s  = blockIdx.x;
  const int hh = blockIdx.y;
  const int b  = blockIdx.z;
  size_t in_idx = ((size_t)(b * Sc + s)) * ((size_t)nh * HDc) + (size_t)hh * HDc + hd;
  float x = (float)P[in_idx];
  float r = x;
  if (apply_rope) {
    float other = (float)P[in_idx + ((hd < 64) ? 64 : -64)];
    float rot   = (hd < 64) ? -other : other;
    size_t cs   = (size_t)(b * Sc + s) * HDc + hd;
    r = x * cosb[cs] + rot * sinb[cs];
  }
  Out[(((size_t)b * nh + hh) * Sc + s) * HDc + hd] = (__bf16)r;
}

// ---------------- Flash attention (online softmax, WMMA) ------------------
// Q: [B,H,S,HD] bf16, K/V: [B,KVH,S,HD] bf16, Ctx: [B,S,H*HD] bf16
// Grid (S/64, H, B), 128 threads = 4 waves; wave w owns q rows q0+16w..+15.
__global__ void __launch_bounds__(128)
attn_kernel(const __bf16* __restrict__ Q, const __bf16* __restrict__ Kr,
            const __bf16* __restrict__ Vr, __bf16* __restrict__ Ctx) {
  const int q0   = blockIdx.x * 64;
  const int h    = blockIdx.y;
  const int b    = blockIdx.z;
  const int kvh  = h / (Hc / KVHc);
  const int tid  = threadIdx.x;
  const int lane = tid & 31;
  const int wid  = tid >> 5;
  const int nl   = lane & 15;
  const int hb   = (lane & 16) ? 8 : 0;
  const float scale = 0.08838834764831845f;     // 1/sqrt(128)

  __shared__ __align__(16) __bf16 Ks[32][136];      // [kv][hd]
  __shared__ __align__(16) __bf16 Vt[HDc][40];      // [hd][kv]  (transposed)
  __shared__ __align__(16) __bf16 Ps[4][16][40];    // per-wave P tile

  // Q fragments for this wave's 16 rows: 4 chunks over HD
  const __bf16* Qb = Q + (((size_t)b * Hc + h) * Sc + q0 + wid * 16) * HDc;
  v16bf qf[4];
#pragma unroll
  for (int kc = 0; kc < 4; ++kc) qf[kc] = frag_a(Qb + kc * 32, HDc, lane);

  v8f o[8] = {};
  float rmax[8], rsum[8];
#pragma unroll
  for (int r = 0; r < 8; ++r) { rmax[r] = -3.0e38f; rsum[r] = 0.f; }

  const __bf16* Kb = Kr + ((size_t)b * KVHc + kvh) * Sc * HDc;
  const __bf16* Vb = Vr + ((size_t)b * KVHc + kvh) * Sc * HDc;
  const int kv_end = q0 + 64;                 // causal upper bound (block-uniform)

  for (int kv0 = 0; kv0 < kv_end; kv0 += 32) {
    // K tile: async B128 straight into LDS. V: register transpose store.
#pragma unroll
    for (int i = 0; i < 4; ++i) {
      int idx = i * 128 + tid;                // 0..511 chunks of 8 bf16
      int row = idx >> 4;
      int col = (idx & 15) * 8;
      async_copy16(Kb + (size_t)(kv0 + row) * HDc + col, &Ks[row][col]);
      uint4 dv = *reinterpret_cast<const uint4*>(Vb + (size_t)(kv0 + row) * HDc + col);
      const __bf16* pv = reinterpret_cast<const __bf16*>(&dv);
#pragma unroll
      for (int e = 0; e < 8; ++e) Vt[col + e][row] = pv[e];
    }
    wait_async();
    __syncthreads();

    // scores: 16x32 per wave, as two 16x16 accumulators
    v8f s[2];
#pragma unroll
    for (int c = 0; c < 2; ++c) {
      s[c] = (v8f){};
#pragma unroll
      for (int kc = 0; kc < 4; ++kc) {
        v16bf bf = frag_b(&Ks[c * 16][kc * 32], 136, lane);
        s[c] = wmma_bf16(qf[kc], bf, s[c]);
      }
      // scale + causal mask
#pragma unroll
      for (int r = 0; r < 8; ++r) {
        int mg = q0 + wid * 16 + r + hb;
        int ng = kv0 + c * 16 + nl;
        s[c][r] = s[c][r] * scale + ((ng > mg) ? NEGC : 0.0f);
      }
    }

    // online softmax: per-row reduce over 16-lane half groups
    float tm[8];
#pragma unroll
    for (int r = 0; r < 8; ++r) tm[r] = fmaxf(s[0][r], s[1][r]);
#pragma unroll
    for (int m = 1; m < 16; m <<= 1)
#pragma unroll
      for (int r = 0; r < 8; ++r) tm[r] = fmaxf(tm[r], __shfl_xor(tm[r], m, 32));

    float corr[8];
#pragma unroll
    for (int r = 0; r < 8; ++r) {
      float nm = fmaxf(rmax[r], tm[r]);
      corr[r]  = __expf(rmax[r] - nm);
      rmax[r]  = nm;
    }
    float ps[8];
#pragma unroll
    for (int r = 0; r < 8; ++r) {
      s[0][r] = __expf(s[0][r] - rmax[r]);
      s[1][r] = __expf(s[1][r] - rmax[r]);
      ps[r]   = s[0][r] + s[1][r];
    }
#pragma unroll
    for (int m = 1; m < 16; m <<= 1)
#pragma unroll
      for (int r = 0; r < 8; ++r) ps[r] += __shfl_xor(ps[r], m, 32);
#pragma unroll
    for (int r = 0; r < 8; ++r) rsum[r] = rsum[r] * corr[r] + ps[r];

    // P (C-layout) -> LDS -> A-layout fragment (per-wave buffer, in-wave RAW)
#pragma unroll
    for (int c = 0; c < 2; ++c)
#pragma unroll
      for (int r = 0; r < 8; ++r)
        Ps[wid][r + hb][c * 16 + nl] = (__bf16)s[c][r];
    asm volatile("s_wait_dscnt 0" ::: "memory");   // DS in-order per wave

    // rescale running O, then O += P @ V
#pragma unroll
    for (int t = 0; t < 8; ++t)
#pragma unroll
      for (int r = 0; r < 8; ++r) o[t][r] *= corr[r];

    v16bf pf = frag_a(&Ps[wid][0][0], 40, lane);
#pragma unroll
    for (int t = 0; t < 8; ++t) {
      v16bf vf = frag_b(&Vt[t * 16][0], 40, lane);  // B[k][n]=V[kv=k][hd=t*16+n]
      o[t] = wmma_bf16(pf, vf, o[t]);
    }
    __syncthreads();
  }

  // epilogue: normalize, write ctx bf16 [B,S,H*HD]
  float inv[8];
#pragma unroll
  for (int r = 0; r < 8; ++r) inv[r] = 1.0f / rsum[r];
  __bf16* Cb = Ctx + ((size_t)b * Sc + q0 + wid * 16) * (Hc * HDc) + (size_t)h * HDc;
#pragma unroll
  for (int t = 0; t < 8; ++t)
#pragma unroll
    for (int r = 0; r < 8; ++r)
      Cb[(size_t)(r + hb) * (Hc * HDc) + t * 16 + nl] = (__bf16)(o[t][r] * inv[r]);
}

// --------------------------------------------------------------------------
extern "C" void kernel_launch(void* const* d_in, const int* in_sizes, int n_in,
                              void* d_out, int out_size, void* d_ws, size_t ws_size,
                              hipStream_t stream) {
  const float* X    = (const float*)d_in[0];  // hidden_states [B,S,D]
  const float* cosb = (const float*)d_in[1];  // [B,S,HD]
  const float* sinb = (const float*)d_in[2];  // [B,S,HD]
  // d_in[3] attention_mask: causal, applied analytically
  const float* Wq = (const float*)d_in[4];    // [H*HD, D]
  const float* Wk = (const float*)d_in[5];    // [KVH*HD, D]
  const float* Wv = (const float*)d_in[6];    // [KVH*HD, D]
  const float* Wo = (const float*)d_in[7];    // [D, H*HD]
  float* out = (float*)d_out;                 // [B,S,D]

  const size_t MS  = (size_t)Bc * Sc;         // 4096 rows
  const size_t NQ  = (size_t)Hc * HDc;        // 4096
  const size_t NKV = (size_t)KVHc * HDc;      // 1024
  const size_t szX = MS * Dc;                 // 16.78M
  const size_t szQ = MS * NQ;                 // 16.78M
  const size_t szK = MS * NKV;                // 4.19M

  // workspace layout, all bf16 (~252 MB)
  __bf16* Xb  = (__bf16*)d_ws;
  __bf16* Wqb = Xb  + szX;
  __bf16* Wkb = Wqb + (size_t)NQ  * Dc;
  __bf16* Wvb = Wkb + (size_t)NKV * Dc;
  __bf16* Wob = Wvb + (size_t)NKV * Dc;
  __bf16* Qpb = Wob + (size_t)Dc  * NQ;
  __bf16* Kpb = Qpb + szQ;
  __bf16* Vpb = Kpb + szK;
  __bf16* Qr  = Vpb + szK;
  __bf16* Kr  = Qr  + szQ;
  __bf16* Vr  = Kr  + szK;
  __bf16* Ctx = Vr  + szK;

  dim3 blk(256);
  // one-pass f32 -> bf16 casts (each thread converts 4 elements)
  cast_bf16_kernel<<<dim3(szX / 1024),              blk, 0, stream>>>(X,  Xb);
  cast_bf16_kernel<<<dim3((NQ  * Dc) / 1024),       blk, 0, stream>>>(Wq, Wqb);
  cast_bf16_kernel<<<dim3((NKV * Dc) / 1024),       blk, 0, stream>>>(Wk, Wkb);
  cast_bf16_kernel<<<dim3((NKV * Dc) / 1024),       blk, 0, stream>>>(Wv, Wvb);
  cast_bf16_kernel<<<dim3((Dc  * NQ) / 1024),       blk, 0, stream>>>(Wo, Wob);

  // Q/K/V projections (bf16 in, bf16 out)
  gemm_wmma<__bf16><<<dim3(MS / 128, NQ  / 64), blk, 0, stream>>>(Xb, Wqb, Qpb, MS, NQ,  Dc);
  gemm_wmma<__bf16><<<dim3(MS / 128, NKV / 64), blk, 0, stream>>>(Xb, Wkb, Kpb, MS, NKV, Dc);
  gemm_wmma<__bf16><<<dim3(MS / 128, NKV / 64), blk, 0, stream>>>(Xb, Wvb, Vpb, MS, NKV, Dc);

  // RoPE + head-major layout
  rope_kernel<<<dim3(Sc, Hc,   Bc), dim3(HDc), 0, stream>>>(Qpb, cosb, sinb, Qr, Hc,   1);
  rope_kernel<<<dim3(Sc, KVHc, Bc), dim3(HDc), 0, stream>>>(Kpb, cosb, sinb, Kr, KVHc, 1);
  rope_kernel<<<dim3(Sc, KVHc, Bc), dim3(HDc), 0, stream>>>(Vpb, cosb, sinb, Vr, KVHc, 0);

  // flash attention
  attn_kernel<<<dim3(Sc / 64, Hc, Bc), dim3(128), 0, stream>>>(Qr, Kr, Vr, Ctx);

  // output projection (bf16 x bf16 -> f32 out)
  gemm_wmma<float><<<dim3(MS / 128, Dc / 64), blk, 0, stream>>>(Ctx, Wob, out, MS, Dc, NQ);
}
